// get_model_27376121545187
// MI455X (gfx1250) — matrix operand order repaired and checked by reading, not compiled, not run on hardware
//
#include <hip/hip_runtime.h>
#include <hip/hip_bf16.h>

typedef __attribute__((ext_vector_type(16))) _Float16 v16h;
typedef __attribute__((ext_vector_type(8)))  float    v8f;

#define NB 8  // batch size

// ----------------------------- helpers -----------------------------------
__global__ void zero_f_kernel(float* __restrict__ p, int n) {
  int t = blockIdx.x * blockDim.x + threadIdx.x;
  if (t < n) p[t] = 0.0f;
}

__global__ void copy_f_kernel(const float* __restrict__ s, float* __restrict__ d, int n) {
  int t = blockIdx.x * blockDim.x + threadIdx.x;
  if (t < n) d[t] = s[t];
}

__global__ void iota_nidx_kernel(int* __restrict__ nidx, int n, int K) {
  int t = blockIdx.x * blockDim.x + threadIdx.x;
  if (t < n) nidx[t] = t % K;
}

// xyz (B,6,N) -> pts (B,N,3), nrm (B,N,3)
__global__ void split_transpose_kernel(const float* __restrict__ xyz,
                                       float* __restrict__ pts,
                                       float* __restrict__ nrm, int N) {
  int t = blockIdx.x * blockDim.x + threadIdx.x;
  if (t >= NB * N) return;
  int b = t / N, n = t % N;
#pragma unroll
  for (int c = 0; c < 3; ++c) {
    pts[(size_t)t * 3 + c] = xyz[((size_t)b * 6 + c) * N + n];
    nrm[(size_t)t * 3 + c] = xyz[((size_t)b * 6 + 3 + c) * N + n];
  }
}

// ----------------------- farthest point sampling --------------------------
// one block (256 threads) per batch; sequential argmax chain matching lax.scan
__global__ __launch_bounds__(256) void fps_kernel(const float* __restrict__ pts,
                                                  int* __restrict__ idx,
                                                  int N, int S) {
  __shared__ float dists[2048];
  __shared__ float rd[256];
  __shared__ int   ri[256];
  __shared__ int   sfar;
  int b = blockIdx.x, tid = threadIdx.x;
  const float* P = pts + (size_t)b * N * 3;
  for (int n = tid; n < N; n += 256) dists[n] = 1e10f;
  int far = 0;
  for (int s = 0; s < S; ++s) {
    if (tid == 0) idx[b * S + s] = far;   // emits previous 'far' (matches scan output)
    float cx = P[far * 3 + 0], cy = P[far * 3 + 1], cz = P[far * 3 + 2];
    __syncthreads();
    float bestd = -1.0f; int besti = 0;
    for (int n = tid; n < N; n += 256) {
      float dx = P[n * 3 + 0] - cx, dy = P[n * 3 + 1] - cy, dz = P[n * 3 + 2] - cz;
      float d = dx * dx + dy * dy + dz * dz;
      d = fminf(dists[n], d);
      dists[n] = d;
      if (d > bestd) { bestd = d; besti = n; }
    }
    rd[tid] = bestd; ri[tid] = besti;
    __syncthreads();
    for (int st = 128; st > 0; st >>= 1) {
      if (tid < st) {
        if (rd[tid + st] > rd[tid] ||
            (rd[tid + st] == rd[tid] && ri[tid + st] < ri[tid])) {
          rd[tid] = rd[tid + st]; ri[tid] = ri[tid + st];
        }
      }
      __syncthreads();
    }
    if (tid == 0) sfar = ri[0];
    __syncthreads();
    far = sfar;
  }
}

__global__ void gather_xyz_kernel(const float* __restrict__ pts,
                                  const int* __restrict__ idx,
                                  float* __restrict__ nx, int N, int S) {
  int t = blockIdx.x * blockDim.x + threadIdx.x;
  if (t >= NB * S) return;
  int b = t / S;
  int p = idx[t];
#pragma unroll
  for (int c = 0; c < 3; ++c) nx[(size_t)t * 3 + c] = pts[((size_t)b * N + p) * 3 + c];
}

// -------------------------- 32-NN selection -------------------------------
__global__ void knn_kernel(const float* __restrict__ pts,
                           const float* __restrict__ q,
                           int* __restrict__ nidx, int N, int S) {
  int t = blockIdx.x * blockDim.x + threadIdx.x;
  if (t >= NB * S) return;
  int b = t / S;
  const float* P = pts + (size_t)b * N * 3;
  float qx = q[t * 3 + 0], qy = q[t * 3 + 1], qz = q[t * 3 + 2];
  float bd[32]; int bi[32];
#pragma unroll
  for (int i = 0; i < 32; ++i) { bd[i] = 3.4e38f; bi[i] = 0; }
  for (int n = 0; n < N; ++n) {
    float dx = P[n * 3 + 0] - qx, dy = P[n * 3 + 1] - qy, dz = P[n * 3 + 2] - qz;
    float d = dx * dx + dy * dy + dz * dz;
    if (d < bd[31]) {                 // strict < keeps lower index on ties (top_k stable)
      int j = 31;
      while (j > 0 && bd[j - 1] > d) { bd[j] = bd[j - 1]; bi[j] = bi[j - 1]; --j; }
      bd[j] = d; bi[j] = n;
    }
  }
#pragma unroll
  for (int i = 0; i < 32; ++i) nidx[(size_t)t * 32 + i] = bi[i];
}

// --------------- relation features + m1 MLP (7->H->1), mean over 9 fps -----
__global__ void relation_w_kernel(const float* __restrict__ pts,
                                  const float* __restrict__ q,
                                  const int* __restrict__ nidx,
                                  const float* __restrict__ W1,  // (7,H)
                                  const float* __restrict__ b1,  // (H)
                                  const float* __restrict__ W2,  // (H,1)
                                  const float* __restrict__ b2,  // (1)
                                  float* __restrict__ w,
                                  int N, int S, int K, int H, float radius) {
  int t = blockIdx.x * blockDim.x + threadIdx.x;
  if (t >= NB * S * K) return;
  int bs = t / K;
  int b = bs / S;
  int pn = nidx[t];
  const float* P = pts + ((size_t)b * N + pn) * 3;
  float rx = P[0] - q[bs * 3 + 0];
  float ry = P[1] - q[bs * 3 + 1];
  float rz = P[2] - q[bs * 3 + 2];
  const float FPx[9] = { 1, 1, 1, 1, -1, -1, -1, -1, 0 };
  const float FPy[9] = { 1, 1, -1, -1, 1, 1, -1, -1, 0 };
  const float FPz[9] = { 1, -1, 1, -1, 1, -1, 1, -1, 0 };
  float acc = 0.0f;
  for (int fp = 0; fp < 9; ++fp) {
    float fx = rx - FPx[fp] * radius;
    float fy = ry - FPy[fp] * radius;
    float fz = rz - FPz[fp] * radius;
    float dist = sqrtf(fx * fx + fy * fy + fz * fz + 1e-12f);
    float f[7] = { rx, ry, rz, fx, fy, fz, dist };
    float o = b2[0];
    for (int j = 0; j < H; ++j) {
      float h = b1[j];
#pragma unroll
      for (int i = 0; i < 7; ++i) h += f[i] * W1[i * H + j];
      h = h > 0.0f ? h : 0.0f;
      o += h * W2[j];
    }
    o = o > 0.0f ? o : 0.0f;
    acc += o;
  }
  w[t] = acc * (1.0f / 9.0f);
}

// pooled[bs, c] = max_k feat[b, nidx[bs,k], c] * w[bs,k]
__global__ void pooled_kernel(const float* __restrict__ feat,
                              const int* __restrict__ nidx,
                              const float* __restrict__ w,
                              float* __restrict__ pooled,
                              int N, int S, int K, int C) {
  int t = blockIdx.x * blockDim.x + threadIdx.x;
  if (t >= NB * S * C) return;
  int c = t % C;
  int bs = t / C;
  int b = bs / S;
  float m = -3.4e38f;
  for (int k = 0; k < K; ++k) {
    int pn = nidx[(size_t)bs * K + k];
    float v = feat[((size_t)b * N + pn) * C + c] * w[(size_t)bs * K + k];
    m = v > m ? v : m;
  }
  pooled[t] = m;
}

// --------------- m2 MLP via WMMA, fully fused -----------------------------
// rows = (b,s,c); hidden[row,k] = relu(x[row]*W1[k]+b1[k]) is generated
// in-register as the WMMA A fragment; out[bs,n] = max_c relu(hidden@W2 + b2)
// accumulated by float-bits atomicMax (valid: ReLU >= 0, out pre-zeroed).
// Out-of-range rows are CLAMPED (not predicated): their D rows are garbage
// but never stored, so loads stay unconditional and fully pipelined.
template <int C, int H1, int H2>
__global__ __launch_bounds__(32) void m2_wmma_kernel(
    const float* __restrict__ x,    // totalRows (pooled, c fastest)
    const float* __restrict__ W1,   // (H1)
    const float* __restrict__ b1,   // (H1)
    const float* __restrict__ W2,   // (H1,H2) row-major
    const float* __restrict__ b2,   // (H2)
    int* __restrict__ out,          // (B*S, H2) float bits, pre-zeroed
    int totalRows) {
  int rowBase = blockIdx.x * 16;
  int nBase   = blockIdx.y * 16;
  int lane = threadIdx.x & 31;
  int half = lane >> 4;
  int l16  = lane & 15;

  int aRow = rowBase + l16;
  float xa = x[(aRow < totalRows) ? aRow : 0];   // clamped, unconditional load
  int nCol = nBase + l16;

  // pull this wave's weight column block toward the WGP ahead of the loads
  __builtin_prefetch(&W2[nBase], 0, 1);

  v8f acc = {};
#pragma unroll
  for (int ks = 0; ks < H1; ks += 32) {
    v16h a, b;
    // A fragment (16x32 f16): half 0 -> K 0..15, half 1 -> K 16..31
    int ka = ks + half * 16;
#pragma unroll
    for (int i = 0; i < 4; ++i) {
      int k0 = ka + 2 * i;
      float h0 = xa * W1[k0] + b1[k0];
      float h1 = xa * W1[k0 + 1] + b1[k0 + 1];
      a[2 * i]     = (_Float16)(h0 > 0.0f ? h0 : 0.0f);
      a[2 * i + 1] = (_Float16)(h1 > 0.0f ? h1 : 0.0f);
      int k2 = ka + 8 + 2 * i;
      float h2 = xa * W1[k2] + b1[k2];
      float h3 = xa * W1[k2 + 1] + b1[k2 + 1];
      a[8 + 2 * i] = (_Float16)(h2 > 0.0f ? h2 : 0.0f);
      a[9 + 2 * i] = (_Float16)(h3 > 0.0f ? h3 : 0.0f);
    }
    // B fragment (32x16 f16): half 0 -> K 0..15, half 1 -> K 16..31
    int kb = ks + half * 16;
#pragma unroll
    for (int i = 0; i < 8; ++i) {
      int k0 = kb + 2 * i;
      b[2 * i]     = (_Float16)W2[(size_t)k0 * H2 + nCol];
      b[2 * i + 1] = (_Float16)W2[(size_t)(k0 + 1) * H2 + nCol];
    }
    acc = __builtin_amdgcn_wmma_f32_16x16x32_f16(false, a, false, b,
                                                 (short)0, acc, false, false);
  }
  float bias = b2[nCol];
#pragma unroll
  for (int v = 0; v < 8; ++v) {
    int row = rowBase + v + half * 8;
    if (row < totalRows) {
      float h = acc[v] + bias;
      h = h > 0.0f ? h : 0.0f;
      int bs = row / C;              // constexpr C -> shift / magic multiply
      atomicMax(&out[(size_t)bs * H2 + nCol], __float_as_int(h));
    }
  }
}

// --------------- fully-connected head via WMMA (M<=16 in one tile) --------
// A rows are clamped to row 0 when aRow >= M: loads are unconditional (no
// exec-mask serialization) and contiguous groups merge into b128 loads;
// garbage rows are masked only at the store.
__global__ __launch_bounds__(32) void fc_wmma_kernel(
    const float* __restrict__ x,     // (M,K)
    const float* __restrict__ W,     // (K,N)
    const float* __restrict__ bias,  // (N)
    const float* __restrict__ g,     // (N) or null
    const float* __restrict__ bb,    // (N) or null
    float* __restrict__ y,           // (M,N)
    int M, int K, int N, int bnrelu) {
  int nBase = blockIdx.x * 16;
  int lane = threadIdx.x & 31;
  int half = lane >> 4;
  int l16  = lane & 15;
  int aRow = l16;
  int nCol = nBase + l16;
  int nColC = (nCol < N) ? nCol : 0;
  const float* xr = x + (size_t)((aRow < M) ? aRow : 0) * K;  // clamped row ptr

  __builtin_prefetch(&W[nBase], 0, 1);

  v8f acc = {};
  for (int ks = 0; ks < K; ks += 32) {
    v16h a, b;
    int ka = ks + half * 16;
#pragma unroll
    for (int i = 0; i < 4; ++i) {
      int k0 = ka + 2 * i;
      a[2 * i]     = (_Float16)xr[k0];
      a[2 * i + 1] = (_Float16)xr[k0 + 1];
      int k2 = ka + 8 + 2 * i;
      a[8 + 2 * i] = (_Float16)xr[k2];
      a[9 + 2 * i] = (_Float16)xr[k2 + 1];
    }
    int kb = ks + half * 16;
#pragma unroll
    for (int i = 0; i < 8; ++i) {
      int k0 = kb + 2 * i;
      b[2 * i]     = (_Float16)W[(size_t)k0 * N + nColC];
      b[2 * i + 1] = (_Float16)W[(size_t)(k0 + 1) * N + nColC];
    }
    acc = __builtin_amdgcn_wmma_f32_16x16x32_f16(false, a, false, b,
                                                 (short)0, acc, false, false);
  }
  if (nCol < N) {
    float bs_ = bias[nCol];
    float gg = bnrelu ? g[nCol] : 1.0f;
    float b2_ = bnrelu ? bb[nCol] : 0.0f;
#pragma unroll
    for (int v = 0; v < 8; ++v) {
      int row = v + half * 8;
      if (row < M) {
        float h = acc[v] + bs_;
        if (bnrelu) { h = h * gg + b2_; h = h > 0.0f ? h : 0.0f; }
        y[(size_t)row * N + nCol] = h;
      }
    }
  }
}

// --------------------------------- host -----------------------------------
extern "C" void kernel_launch(void* const* d_in, const int* in_sizes, int n_in,
                              void* d_out, int out_size, void* d_ws, size_t ws_size,
                              hipStream_t stream) {
  (void)in_sizes; (void)n_in; (void)out_size; (void)ws_size;
  const float* xyz = (const float*)d_in[0];
  auto Pm = [&](int i) { return (const float*)d_in[i]; };

  char* ws = (char*)d_ws;
  size_t off = 0;
  auto allocF = [&](size_t n) { float* p = (float*)(ws + off); off += n * sizeof(float); return p; };
  auto allocI = [&](size_t n) { int*   p = (int*)(ws + off);   off += n * sizeof(int);   return p; };

  const int Ns[6] = { 2048, 512, 256, 128, 32, 1 };
  const int Cs[6] = { 3, 128, 256, 256, 512, 1024 };

  float* xyzL[6]; float* featL[6];
  xyzL[0]  = allocF((size_t)NB * 2048 * 3);
  featL[0] = allocF((size_t)NB * 2048 * 3);
  for (int l = 1; l <= 5; ++l) {
    xyzL[l]  = allocF((size_t)NB * Ns[l] * 3);
    featL[l] = allocF((size_t)NB * Ns[l] * Cs[l]);
  }
  float* pooled = allocF((size_t)NB * 512 * 64);   // 262144 floats = max B*S*Cin
  float* wbuf   = allocF((size_t)NB * 512 * 32);
  float* fca    = allocF((size_t)NB * 512);
  float* fcb    = allocF((size_t)NB * 256);
  int* fpsidx   = allocI((size_t)NB * 512);
  int* nidx     = allocI((size_t)NB * 512 * 32);

  split_transpose_kernel<<<(NB * 2048 + 255) / 256, 256, 0, stream>>>(xyz, xyzL[0], featL[0], 2048);

  const int Hm1[5]   = { 32, 32, 64, 64, 128 };
  const int H2s[5]   = { 128, 256, 256, 512, 1024 };
  const float rad[5] = { 0.15f, 0.25f, 0.4f, 0.6f, 10.0f };
  const int K = 32;

  for (int l = 0; l < 5; ++l) {
    int N = Ns[l], S = Ns[l + 1], Cin = Cs[l], H2 = H2s[l];
    const float* pin = xyzL[l];
    const float* fin = featL[l];
    float* nx = xyzL[l + 1];
    float* fout = featL[l + 1];
    int base = 1 + l * 8;  // conv params: m1W1,m1b1,m1W2,m1b2,m2W1,m2b1,m2W2,m2b2

    if (l < 4) {
      fps_kernel<<<NB, 256, 0, stream>>>(pin, fpsidx, N, S);
      gather_xyz_kernel<<<(NB * S + 255) / 256, 256, 0, stream>>>(pin, fpsidx, nx, N, S);
      knn_kernel<<<(NB * S + 255) / 256, 256, 0, stream>>>(pin, nx, nidx, N, S);
    } else {
      zero_f_kernel<<<1, 64, 0, stream>>>(nx, NB * 3);
      iota_nidx_kernel<<<1, 256, 0, stream>>>(nidx, NB * K, K);
    }

    relation_w_kernel<<<(NB * S * K + 255) / 256, 256, 0, stream>>>(
        pin, nx, nidx, Pm(base + 0), Pm(base + 1), Pm(base + 2), Pm(base + 3),
        wbuf, N, S, K, Hm1[l], rad[l]);

    pooled_kernel<<<(NB * S * Cin + 255) / 256, 256, 0, stream>>>(
        fin, nidx, wbuf, pooled, N, S, K, Cin);

    int rows = NB * S * Cin;
    zero_f_kernel<<<(NB * S * H2 + 255) / 256, 256, 0, stream>>>(fout, NB * S * H2);
    dim3 grid((rows + 15) / 16, H2 / 16);
    const float* mW1 = Pm(base + 4); const float* mb1 = Pm(base + 5);
    const float* mW2 = Pm(base + 6); const float* mb2 = Pm(base + 7);
    switch (l) {
      case 0: m2_wmma_kernel<3,   64,  128><<<grid, 32, 0, stream>>>(pooled, mW1, mb1, mW2, mb2, (int*)fout, rows); break;
      case 1: m2_wmma_kernel<128, 64,  256><<<grid, 32, 0, stream>>>(pooled, mW1, mb1, mW2, mb2, (int*)fout, rows); break;
      case 2: m2_wmma_kernel<256, 128, 256><<<grid, 32, 0, stream>>>(pooled, mW1, mb1, mW2, mb2, (int*)fout, rows); break;
      case 3: m2_wmma_kernel<256, 128, 512><<<grid, 32, 0, stream>>>(pooled, mW1, mb1, mW2, mb2, (int*)fout, rows); break;
      case 4: m2_wmma_kernel<512, 256, 1024><<<grid, 32, 0, stream>>>(pooled, mW1, mb1, mW2, mb2, (int*)fout, rows); break;
    }
  }

  // classifier head: fc1(1024->512)+bn+relu, fc2(512->256)+bn+relu, fc3(256->40)
  fc_wmma_kernel<<<512 / 16, 32, 0, stream>>>(featL[5], Pm(41), Pm(42), Pm(43), Pm(44),
                                              fca, NB, 1024, 512, 1);
  fc_wmma_kernel<<<256 / 16, 32, 0, stream>>>(fca, Pm(45), Pm(46), Pm(47), Pm(48),
                                              fcb, NB, 512, 256, 1);
  fc_wmma_kernel<<<(40 + 15) / 16, 32, 0, stream>>>(fcb, Pm(49), Pm(50), nullptr, nullptr,
                                                    (float*)d_out, NB, 256, 40, 0);

  // second output: s_list[2] = level-3 centers (8,128,3)
  copy_f_kernel<<<(NB * 128 * 3 + 255) / 256, 256, 0, stream>>>(
      xyzL[3], (float*)d_out + NB * 40, NB * 128 * 3);
}
